// SelectiveSSM_89945205113174
// MI455X (gfx1250) — matrix operand order repaired and checked by reading, compiled, not run on hardware
//
#include <hip/hip_runtime.h>

// ---------------- problem constants ----------------
#define D_MODEL 1024
#define D_STATE 16
#define D_CONV  4
#define D_INNER 2048
#define BATCH   4
#define SEQLEN  4096
#define MROWS   (BATCH*SEQLEN)      // 16384 rows for all GEMMs
#define NCHAN   (BATCH*D_INNER)     // 8192 scan channels
#define CHUNKS  16                  // L-chunking for parallel linear-recurrence scan
#define CLEN    (SEQLEN/CHUNKS)     // 256
#define CONV_TT 128                 // timesteps per conv thread

#define LOG2E 1.4426950408889634f
#define LN2   0.6931471805599453f

typedef __bf16 v16bf  __attribute__((ext_vector_type(16)));
typedef __bf16 bf16x8 __attribute__((ext_vector_type(8)));
typedef float  v8f    __attribute__((ext_vector_type(8)));

__device__ __forceinline__ float softplus_f(float v) {
  float e = __builtin_amdgcn_exp2f(-fabsf(v) * LOG2E);
  return fmaxf(v, 0.f) + LN2 * __builtin_amdgcn_logf(1.f + e);
}
__device__ __forceinline__ float silu_f(float v) {
  float e = __builtin_amdgcn_exp2f(-v * LOG2E);
  return v * __builtin_amdgcn_rcpf(1.f + e);
}

// ---------------------------------------------------------------------------
// Elementwise f32 -> bf16 (8 elems / thread, b128 in, b128 out)
// ---------------------------------------------------------------------------
__global__ __launch_bounds__(256)
void cvt_f32_bf16(const float* __restrict__ src, __bf16* __restrict__ dst) {
  const size_t i = ((size_t)blockIdx.x * 256 + threadIdx.x) * 8;
  float4 v0 = *(const float4*)(src + i);
  float4 v1 = *(const float4*)(src + i + 4);
  bf16x8 o;
  o[0] = (__bf16)v0.x; o[1] = (__bf16)v0.y; o[2] = (__bf16)v0.z; o[3] = (__bf16)v0.w;
  o[4] = (__bf16)v1.x; o[5] = (__bf16)v1.y; o[6] = (__bf16)v1.z; o[7] = (__bf16)v1.w;
  *(bf16x8*)(dst + i) = o;
}

// ---------------------------------------------------------------------------
// GEMM: out[M,N] = A[M,K] @ W[N,K]^T, bf16 inputs, f32 WMMA accumulate.
// One wave -> 16(M) x (16*NT)(N) strip; K loop software-pipelined 2-deep
// (K % 64 == 0 for all calls) so slab N+1 loads overlap slab N WMMAs.
// EPI: 0 = none, 1 = softplus(acc + bias[n])
// ---------------------------------------------------------------------------
template<int NT> struct Frag {
  bf16x8 a0, a1;          // A 16x32: lane<16 -> K {0..7,16..23}; lane>=16 -> {8..15,24..31}
  bf16x8 b0[NT], b1[NT];  // B=W^T:   lane<16 -> K {0..7,16..23}; lane>=16 -> {8..15,24..31}
};

template<int NT, int EPI>
__global__ __launch_bounds__(256)
void gemm_bf16_wmma(const __bf16* __restrict__ A, const __bf16* __restrict__ W,
                    const float* __restrict__ bias, float* __restrict__ out,
                    int M, int N, int K) {
  const int lane  = threadIdx.x & 31;
  const int gwave = (blockIdx.x << 3) + (threadIdx.x >> 5);
  const int nGrp  = N / (16 * NT);
  const int tm    = gwave / nGrp;
  const int tn    = gwave - tm * nGrp;
  if (tm >= (M >> 4)) return;
  const int m0 = tm << 4;
  const int n0 = tn * (16 * NT);
  const int lh = lane >> 4;
  const int ll = lane & 15;

  const __bf16* pa = A + (size_t)(m0 + ll) * K + (lh << 3);
  const __bf16* pw[NT];
  #pragma unroll
  for (int nt = 0; nt < NT; ++nt)
    pw[nt] = W + (size_t)(n0 + nt * 16 + ll) * K + (lh << 3);

  v8f acc[NT];
  #pragma unroll
  for (int i = 0; i < NT; ++i)
    #pragma unroll
    for (int j = 0; j < 8; ++j) acc[i][j] = 0.f;

  auto load_frag = [&](Frag<NT>& f, int k0) {
    f.a0 = *(const bf16x8*)(pa + k0);
    f.a1 = *(const bf16x8*)(pa + k0 + 16);
    #pragma unroll
    for (int nt = 0; nt < NT; ++nt) {
      f.b0[nt] = *(const bf16x8*)(pw[nt] + k0);
      f.b1[nt] = *(const bf16x8*)(pw[nt] + k0 + 16);
    }
  };
  auto compute = [&](const Frag<NT>& f) {
    v16bf af = __builtin_shufflevector(f.a0, f.a1,
        0,1,2,3,4,5,6,7,8,9,10,11,12,13,14,15);
    #pragma unroll
    for (int nt = 0; nt < NT; ++nt) {
      v16bf bv = __builtin_shufflevector(f.b0[nt], f.b1[nt],
          0,1,2,3,4,5,6,7,8,9,10,11,12,13,14,15);
      acc[nt] = __builtin_amdgcn_wmma_f32_16x16x32_bf16(
          false, af, false, bv, (short)0, acc[nt], false, false);
    }
  };

  Frag<NT> f0, f1;
  load_frag(f0, 0);
  for (int k0 = 0; k0 < K; k0 += 64) {
    load_frag(f1, k0 + 32);        // in flight while f0 computes
    compute(f0);
    if (k0 + 64 < K) load_frag(f0, k0 + 64);
    compute(f1);
  }

  // C/D layout: VGPR j -> M = m0 + j + 8*lh, N = n0 + nt*16 + ll
  #pragma unroll
  for (int nt = 0; nt < NT; ++nt) {
    const int n = n0 + nt * 16 + ll;
    float bv = 0.f;
    if (EPI == 1) bv = bias[n];
    #pragma unroll
    for (int j = 0; j < 8; ++j) {
      const int m = m0 + (lh << 3) + j;
      float v = acc[nt][j];
      if (EPI == 1) v = softplus_f(v + bv);
      out[(size_t)m * N + n] = v;
    }
  }
}

// ---------------------------------------------------------------------------
// Causal depthwise conv1d (k=4) + bias + SiLU. One thread: one channel,
// CONV_TT timesteps with rolling registers. Coalesced across channels.
// ---------------------------------------------------------------------------
__global__ __launch_bounds__(256)
void conv_silu_kernel(const float* __restrict__ xin, const float* __restrict__ cw,
                      const float* __restrict__ cb, float* __restrict__ xc) {
  const int c  = blockIdx.x * 256 + threadIdx.x;
  const int b  = blockIdx.z;
  const int t0 = blockIdx.y * CONV_TT;
  const float w0 = cw[c * 4 + 0], w1 = cw[c * 4 + 1];
  const float w2 = cw[c * 4 + 2], w3 = cw[c * 4 + 3];
  const float bias = cb[c];
  const float* base = xin + (size_t)b * SEQLEN * D_INNER + c;
  float x0 = (t0 >= 3) ? base[(size_t)(t0 - 3) * D_INNER] : 0.f;
  float x1 = (t0 >= 2) ? base[(size_t)(t0 - 2) * D_INNER] : 0.f;
  float x2 = (t0 >= 1) ? base[(size_t)(t0 - 1) * D_INNER] : 0.f;
  for (int t = t0; t < t0 + CONV_TT; ++t) {
    float x3  = base[(size_t)t * D_INNER];
    float acc = fmaf(w0, x0, fmaf(w1, x1, fmaf(w2, x2, fmaf(w3, x3, bias))));
    xc[((size_t)b * SEQLEN + t) * D_INNER + c] = silu_f(acc);
    x0 = x1; x1 = x2; x2 = x3;
  }
}

// ---------------------------------------------------------------------------
// Chunked parallel linear-recurrence scan: h_t = a_t*h_{t-1} + dt*B_t*x_t
// pass1: per (chunk,channel): Aprod[s] = prod(a), Hend[s] = local scan (h_in=0)
// pass2: sequential combine over 16 chunks -> Hin per chunk
// pass3: replay with correct h_in, emit y_t = <h_t,C_t> + D*x_t as bf16
// ---------------------------------------------------------------------------
__global__ __launch_bounds__(256)
void scan_pass1(const float* __restrict__ xc, const float* __restrict__ dt,
                const float* __restrict__ Bp, const float* __restrict__ A_log,
                float* __restrict__ Aprod, float* __restrict__ Hend) {
  const int chan  = blockIdx.x * 256 + threadIdx.x;
  const int chunk = blockIdx.y;
  const int b = chan >> 11;
  const int c = chan & (D_INNER - 1);

  float A2[16], ap[16], h[16];
  {
    const float4* al4 = (const float4*)(A_log + c * D_STATE);
    float al[16];
    #pragma unroll
    for (int q = 0; q < 4; ++q) *(float4*)&al[4 * q] = al4[q];
    #pragma unroll
    for (int s = 0; s < 16; ++s) {
      A2[s] = -__builtin_amdgcn_exp2f(al[s] * LOG2E) * LOG2E;
      ap[s] = 1.f; h[s] = 0.f;
    }
  }
  const int t0 = chunk * CLEN;
  const float* dtp = dt + (size_t)b * SEQLEN * D_INNER + c;
  const float* xcp = xc + (size_t)b * SEQLEN * D_INNER + c;
  for (int t = t0; t < t0 + CLEN; ++t) {
    const size_t idx = (size_t)t * D_INNER;
    const float d = dtp[idx];
    const float x = xcp[idx];
    const float4* B4 = (const float4*)(Bp + ((size_t)b * SEQLEN + t) * D_STATE);
    float Bt[16];
    #pragma unroll
    for (int q = 0; q < 4; ++q) *(float4*)&Bt[4 * q] = B4[q];
    const float dx = d * x;
    #pragma unroll
    for (int s = 0; s < 16; ++s) {
      float a = __builtin_amdgcn_exp2f(d * A2[s]);
      h[s]  = fmaf(a, h[s], dx * Bt[s]);
      ap[s] *= a;
    }
  }
  const size_t off = ((size_t)chunk * NCHAN + chan) * 16;
  #pragma unroll
  for (int q = 0; q < 4; ++q) {
    ((float4*)(Aprod + off))[q] = make_float4(ap[4*q], ap[4*q+1], ap[4*q+2], ap[4*q+3]);
    ((float4*)(Hend  + off))[q] = make_float4(h[4*q],  h[4*q+1],  h[4*q+2],  h[4*q+3]);
  }
}

__global__ __launch_bounds__(256)
void scan_pass2(const float* __restrict__ Aprod, const float* __restrict__ Hend,
                float* __restrict__ Hin) {
  const int chan = blockIdx.x * 256 + threadIdx.x;
  float h[16];
  #pragma unroll
  for (int s = 0; s < 16; ++s) h[s] = 0.f;
  for (int ch = 0; ch < CHUNKS; ++ch) {
    const size_t off = ((size_t)ch * NCHAN + chan) * 16;
    #pragma unroll
    for (int q = 0; q < 4; ++q)
      ((float4*)(Hin + off))[q] = make_float4(h[4*q], h[4*q+1], h[4*q+2], h[4*q+3]);
    float ap[16], he[16];
    #pragma unroll
    for (int q = 0; q < 4; ++q) {
      *(float4*)&ap[4 * q] = ((const float4*)(Aprod + off))[q];
      *(float4*)&he[4 * q] = ((const float4*)(Hend  + off))[q];
    }
    #pragma unroll
    for (int s = 0; s < 16; ++s) h[s] = fmaf(ap[s], h[s], he[s]);
  }
}

__global__ __launch_bounds__(256)
void scan_pass3(const float* __restrict__ xc, const float* __restrict__ dt,
                const float* __restrict__ Bp, const float* __restrict__ Cp,
                const float* __restrict__ A_log, const float* __restrict__ Dp,
                const float* __restrict__ Hin, __bf16* __restrict__ ybf) {
  const int chan  = blockIdx.x * 256 + threadIdx.x;
  const int chunk = blockIdx.y;
  const int b = chan >> 11;
  const int c = chan & (D_INNER - 1);

  float A2[16], h[16];
  {
    const float4* al4 = (const float4*)(A_log + c * D_STATE);
    float al[16];
    #pragma unroll
    for (int q = 0; q < 4; ++q) *(float4*)&al[4 * q] = al4[q];
    const size_t off = ((size_t)chunk * NCHAN + chan) * 16;
    #pragma unroll
    for (int q = 0; q < 4; ++q) *(float4*)&h[4 * q] = ((const float4*)(Hin + off))[q];
    #pragma unroll
    for (int s = 0; s < 16; ++s)
      A2[s] = -__builtin_amdgcn_exp2f(al[s] * LOG2E) * LOG2E;
  }
  const float Dc = Dp[c];
  const int t0 = chunk * CLEN;
  const float* dtp = dt + (size_t)b * SEQLEN * D_INNER + c;
  const float* xcp = xc + (size_t)b * SEQLEN * D_INNER + c;
  __bf16* yp = ybf + (size_t)b * SEQLEN * D_INNER + c;
  for (int t = t0; t < t0 + CLEN; ++t) {
    const size_t idx = (size_t)t * D_INNER;
    const float d = dtp[idx];
    const float x = xcp[idx];
    const size_t bc = ((size_t)b * SEQLEN + t) * D_STATE;
    float Bt[16], Ct[16];
    #pragma unroll
    for (int q = 0; q < 4; ++q) {
      *(float4*)&Bt[4 * q] = ((const float4*)(Bp + bc))[q];
      *(float4*)&Ct[4 * q] = ((const float4*)(Cp + bc))[q];
    }
    const float dx = d * x;
    float y = Dc * x;
    #pragma unroll
    for (int s = 0; s < 16; ++s) {
      float a = __builtin_amdgcn_exp2f(d * A2[s]);
      h[s] = fmaf(a, h[s], dx * Bt[s]);
      y = fmaf(h[s], Ct[s], y);
    }
    yp[idx] = (__bf16)y;
  }
}

// ---------------------------------------------------------------------------
extern "C" void kernel_launch(void* const* d_in, const int* in_sizes, int n_in,
                              void* d_out, int out_size, void* d_ws, size_t ws_size,
                              hipStream_t stream) {
  (void)in_sizes; (void)n_in; (void)out_size; (void)ws_size;
  const float* x          = (const float*)d_in[0];
  const float* x_proj_w   = (const float*)d_in[1];
  const float* dt_proj_w  = (const float*)d_in[2];
  const float* dt_proj_b  = (const float*)d_in[3];
  const float* A_log      = (const float*)d_in[4];
  const float* Dp         = (const float*)d_in[5];
  const float* B_proj_w   = (const float*)d_in[6];
  const float* C_proj_w   = (const float*)d_in[7];
  const float* conv_w     = (const float*)d_in[8];
  const float* conv_b     = (const float*)d_in[9];
  const float* out_proj_w = (const float*)d_in[10];
  float* out = (float*)d_out;

  // workspace carving
  char* ws = (char*)d_ws;
  const size_t SZ_INNER   = (size_t)MROWS * D_INNER * sizeof(float);     // 128 MiB
  const size_t SZ_BC      = (size_t)MROWS * D_STATE * sizeof(float);     // 1 MiB
  const size_t SZ_ST      = (size_t)CHUNKS * NCHAN * 16 * sizeof(float); // 8 MiB
  const size_t SZ_XBF     = (size_t)MROWS * D_MODEL * sizeof(__bf16);    // 32 MiB
  const size_t SZ_YBF     = (size_t)MROWS * D_INNER * sizeof(__bf16);    // 64 MiB
  const size_t SZ_WBIG    = (size_t)D_INNER * D_MODEL * sizeof(__bf16);  // 4 MiB
  const size_t SZ_WSMALL  = (size_t)D_STATE * D_MODEL * sizeof(__bf16);  // 32 KiB
  float*  x_inner = (float*) ws;  ws += SZ_INNER;
  float*  delta   = (float*) ws;  ws += SZ_INNER;
  float*  xc      = (float*) ws;  ws += SZ_INNER;
  float*  Bb      = (float*) ws;  ws += SZ_BC;
  float*  Cc      = (float*) ws;  ws += SZ_BC;
  float*  Aprod   = (float*) ws;  ws += SZ_ST;
  float*  Hend    = (float*) ws;  ws += SZ_ST;
  float*  Hin     = (float*) ws;  ws += SZ_ST;
  __bf16* xbf     = (__bf16*)ws;  ws += SZ_XBF;
  __bf16* ybf     = (__bf16*)ws;  ws += SZ_YBF;
  __bf16* wxbf    = (__bf16*)ws;  ws += SZ_WBIG;
  __bf16* wdtbf   = (__bf16*)ws;  ws += SZ_WBIG;
  __bf16* wobf    = (__bf16*)ws;  ws += SZ_WBIG;
  __bf16* wBbf    = (__bf16*)ws;  ws += SZ_WSMALL;
  __bf16* wCbf    = (__bf16*)ws;  ws += SZ_WSMALL;

  // 0) one-shot f32 -> bf16 conversions (GEMM operands)
  cvt_f32_bf16<<<(MROWS * D_MODEL) / 2048, 256, 0, stream>>>(x, xbf);
  cvt_f32_bf16<<<(D_INNER * D_MODEL) / 2048, 256, 0, stream>>>(x_proj_w, wxbf);
  cvt_f32_bf16<<<(D_INNER * D_MODEL) / 2048, 256, 0, stream>>>(dt_proj_w, wdtbf);
  cvt_f32_bf16<<<(D_MODEL * D_INNER) / 2048, 256, 0, stream>>>(out_proj_w, wobf);
  cvt_f32_bf16<<<(D_STATE * D_MODEL) / 2048, 256, 0, stream>>>(B_proj_w, wBbf);
  cvt_f32_bf16<<<(D_STATE * D_MODEL) / 2048, 256, 0, stream>>>(C_proj_w, wCbf);

  // 1) x_inner = x @ x_proj_w^T          (M=16384, K=1024, N=2048)
  {
    int M = MROWS, N = D_INNER, K = D_MODEL;
    int blocks = ((M / 16) * (N / 64)) / 8;
    gemm_bf16_wmma<4, 0><<<blocks, 256, 0, stream>>>(xbf, wxbf, nullptr, x_inner, M, N, K);
  }
  // 2) delta = softplus(x @ dt_proj_w^T + dt_proj_b)
  {
    int M = MROWS, N = D_INNER, K = D_MODEL;
    int blocks = ((M / 16) * (N / 64)) / 8;
    gemm_bf16_wmma<4, 1><<<blocks, 256, 0, stream>>>(xbf, wdtbf, dt_proj_b, delta, M, N, K);
  }
  // 3) B = x @ B_proj_w^T ; 4) C = x @ C_proj_w^T   (N=16 -> NT=1)
  {
    int M = MROWS, N = D_STATE, K = D_MODEL;
    int blocks = ((M / 16) * (N / 16)) / 8;
    gemm_bf16_wmma<1, 0><<<blocks, 256, 0, stream>>>(xbf, wBbf, nullptr, Bb, M, N, K);
    gemm_bf16_wmma<1, 0><<<blocks, 256, 0, stream>>>(xbf, wCbf, nullptr, Cc, M, N, K);
  }
  // 5) xc = silu(causal depthwise conv(x_inner) + conv_b)
  {
    dim3 grid(D_INNER / 256, SEQLEN / CONV_TT, BATCH);
    conv_silu_kernel<<<grid, 256, 0, stream>>>(x_inner, conv_w, conv_b, xc);
  }
  // 6) chunked selective scan; pass3 emits y directly as bf16
  {
    dim3 g1(NCHAN / 256, CHUNKS);
    scan_pass1<<<g1, 256, 0, stream>>>(xc, delta, Bb, A_log, Aprod, Hend);
    scan_pass2<<<NCHAN / 256, 256, 0, stream>>>(Aprod, Hend, Hin);
    scan_pass3<<<g1, 256, 0, stream>>>(xc, delta, Bb, Cc, A_log, Dp, Hin, ybf);
  }
  // 7) out = y @ out_proj_w^T            (M=16384, K=2048, N=1024)
  {
    int M = MROWS, N = D_MODEL, K = D_INNER;
    int blocks = ((M / 16) * (N / 64)) / 8;
    gemm_bf16_wmma<4, 0><<<blocks, 256, 0, stream>>>(ybf, wobf, nullptr, out, M, N, K);
  }
}
